// CliffordFourierHead_68186900791669
// MI455X (gfx1250) — matrix operand order repaired and checked by reading, not compiled, not run on hardware
//
#include <hip/hip_runtime.h>
#include <math.h>

// ---------------- problem dims (fixed by the reference) ----------------
#define BB    1024
#define CIN   256
#define CHID  512
#define COUT  128
#define NPATH 20
#define INV_SQRT2 0.70710678118654752440f
#define EPSN  1e-6f
#define LDSA  40   // padded LDS row stride for A tiles (bf16 elems)

typedef __bf16          v16bf __attribute__((ext_vector_type(16)));
typedef float           v8f   __attribute__((ext_vector_type(8)));
typedef float           v4f   __attribute__((ext_vector_type(4)));
typedef unsigned int    v8u   __attribute__((ext_vector_type(8)));
typedef unsigned short  v8us  __attribute__((ext_vector_type(8)));

// ---------------- Cl(3,0) tables, blades [1,e1,e2,e3,e12,e13,e23,e123] ----
// term encoding: i | (k<<3) | (neg<<6)  meaning  sign * x[i] * xr[k]
#define E(i,k,n) ((i) | ((k)<<3) | ((n)<<6))
__device__ const int GRADE8[8]   = {0,1,1,1,2,2,2,3};
__device__ const int GP_NSLOT[8] = {4,6,6,6,6,6,6,4};
__device__ const int GP_T[8][6] = {
  {0,4,10,16, 0, 0},
  {1,5, 6,11,12,17},
  {1,5, 6,11,12,17},
  {1,5, 6,11,12,17},
  {2,7, 8,13,14,18},
  {2,7, 8,13,14,18},
  {2,7, 8,13,14,18},
  {3,9,15,19, 0, 0}};
__device__ const int GP_NTERM[8][6] = {
  {1,3,3,1,0,0},
  {1,1,2,2,1,1},
  {1,1,2,2,1,1},
  {1,1,2,2,1,1},
  {1,2,1,1,2,1},
  {1,2,1,1,2,1},
  {1,2,1,1,2,1},
  {1,3,3,1,0,0}};
__device__ const int GP_TERM[8][6][3] = {
  { {E(0,0,0),0,0}, {E(1,1,0),E(2,2,0),E(3,3,0)}, {E(4,4,1),E(5,5,1),E(6,6,1)}, {E(7,7,1),0,0}, {0,0,0}, {0,0,0} },
  { {E(0,1,0),0,0}, {E(1,0,0),0,0}, {E(2,4,1),E(3,5,1),0}, {E(4,2,0),E(5,3,0),0}, {E(6,7,1),0,0}, {E(7,6,1),0,0} },
  { {E(0,2,0),0,0}, {E(2,0,0),0,0}, {E(1,4,0),E(3,6,1),0}, {E(4,1,1),E(6,3,0),0}, {E(5,7,0),0,0}, {E(7,5,0),0,0} },
  { {E(0,3,0),0,0}, {E(3,0,0),0,0}, {E(1,5,0),E(2,6,0),0}, {E(5,1,1),E(6,2,1),0}, {E(4,7,1),0,0}, {E(7,4,0),0,0} },
  { {E(0,4,0),0,0}, {E(1,2,0),E(2,1,1),0}, {E(3,7,0),0,0}, {E(4,0,0),0,0}, {E(5,6,1),E(6,5,0),0}, {E(7,3,0),0,0} },
  { {E(0,5,0),0,0}, {E(1,3,0),E(3,1,1),0}, {E(2,7,1),0,0}, {E(5,0,0),0,0}, {E(4,6,0),E(6,4,1),0}, {E(7,2,1),0,0} },
  { {E(0,6,0),0,0}, {E(2,3,0),E(3,2,1),0}, {E(1,7,0),0,0}, {E(6,0,0),0,0}, {E(4,5,1),E(5,4,0),0}, {E(7,1,0),0,0} },
  { {E(0,7,0),0,0}, {E(1,6,0),E(2,5,1),E(3,4,0)}, {E(6,1,0),E(5,2,1),E(4,3,0)}, {E(7,0,0),0,0}, {0,0,0}, {0,0,0} }};

// ---------------- helpers ----------------
__device__ inline unsigned short f2bf(float f) {  // f32 -> bf16 RNE
  union { float f; unsigned u; } c; c.f = f;
  unsigned r = c.u + 0x7FFFu + ((c.u >> 16) & 1u);
  return (unsigned short)(r >> 16);
}
__device__ inline float sigmoidf(float x) { return 1.f / (1.f + __expf(-x)); }

__device__ inline float gp_slot_val(int j, int s, const float* a, const float* r) {
  float v = 0.f;
  const int nt = GP_NTERM[j][s];
  for (int q = 0; q < nt; ++q) {
    const int e = GP_TERM[j][s][q];
    const float p = a[e & 7] * r[(e >> 3) & 7];
    v += (e & 64) ? -p : p;
  }
  return v;
}

// async global->LDS 16B copy (GLOBAL_LOAD_ASYNC_TO_LDS_B128, ASYNCcnt tracked)
__device__ inline void async_copy16(unsigned lds_byte_addr, const void* gptr) {
  asm volatile("global_load_async_to_lds_b128 %0, %1, off"
               :: "v"(lds_byte_addr), "v"((unsigned long long)(size_t)gptr)
               : "memory");
}
__device__ inline void wait_async0() {
  asm volatile("s_wait_asynccnt 0x0" ::: "memory");
}
__device__ inline unsigned lds_addr_of(const unsigned short* p) {
  return (unsigned)(size_t)p;   // low 32 bits of flat LDS address == LDS byte offset
}

// WMMA fragment loads (wave32, 16-bit A 16x32 / B 32x16 layouts per CDNA5 ISA 7.12.2)
__device__ inline v16bf load_frag_a(const unsigned short* As, int rowBase, int l2, int hi) {
  const unsigned short* rp = As + (rowBase + l2) * LDSA;
  v8u t;
#pragma unroll
  for (int v = 0; v < 8; ++v) {
    const int k = (v < 4 ? (v << 1) : 16 + ((v - 4) << 1)) + (hi << 3);
    t[v] = *(const unsigned*)(rp + k);
  }
  return __builtin_bit_cast(v16bf, t);
}
// B tile in LDS: row m (128 rows), 32 bf16 per row, granule-XOR swizzled
__device__ inline v16bf load_frag_b(const unsigned short* Bs, int colBase, int l2, int hi) {
  const int m = colBase + l2;
  const unsigned short* cp = Bs + m * 32;
  const int sw = (m >> 2) & 3;
  v8u t;
#pragma unroll
  for (int v = 0; v < 8; ++v) {
    const int k = (v < 4 ? (v << 1) : 16 + ((v - 4) << 1)) + (hi << 3);
    const int o = (k & 7) + ((((k >> 3) ^ sw)) << 3);
    t[v] = *(const unsigned*)(cp + o);
  }
  return __builtin_bit_cast(v16bf, t);
}

// ---------------- prepack kernels ----------------
// x:[BK][8] f32 -> dst:[8][BK] bf16 (blade-major so GEMM A-panels are contiguous)
__global__ void pack_bf16_bm_kernel(const float* __restrict__ src, unsigned short* __restrict__ dst, int BK) {
  int i = blockIdx.x * 256 + threadIdx.x;
  if (i >= 8 * BK) return;
  const int bk = i % BK, j = i / BK;
  dst[i] = f2bf(src[(size_t)bk * 8 + j]);
}

// w:[M][K][4] -> WT:[8][K/32][M][32] (K-tiled, transposed, m-major rows of 32)
__global__ void pack_wlin_kernel(const float* __restrict__ w, unsigned short* __restrict__ WT, int K, int M) {
  int i = blockIdx.x * 256 + threadIdx.x;
  if (i >= 8 * K * M) return;
  const int nKt = K >> 5;
  const int kin = i & 31;
  int r = i >> 5;
  const int m = r % M; r /= M;
  const int kt = r % nKt;
  const int j = r / nKt;
  const int k = (kt << 5) + kin;
  WT[i] = f2bf(w[((size_t)m * K + k) * 4 + GRADE8[j]]);
}

// fused 8-slot weights, tiled: WT:[8][(K*8)/32][M][32]; K' = n*8+s
__global__ void pack_wp_kernel(const float* __restrict__ wlin, const float* __restrict__ wgp,
                               unsigned short* __restrict__ WT, int K, int M) {
  int i = blockIdx.x * 256 + threadIdx.x;
  if (i >= 8 * K * 8 * M) return;
  const int nKt = K >> 2;            // (8K)/32
  const int kin = i & 31;
  int r = i >> 5;
  const int m = r % M; r /= M;
  const int kt = r % nKt;
  const int j = r / nKt;
  const int kp = (kt << 5) + kin;    // fused K' index
  const int n = kp >> 3, s = kp & 7;
  const int ns = GP_NSLOT[j];
  float v = 0.f;
  if (s < ns)       v = wgp [((size_t)m * K + n) * NPATH + GP_T[j][s]];
  else if (s == ns) v = wlin[((size_t)m * K + n) * 4 + GRADE8[j]];
  WT[i] = f2bf(v);
}

// ---------------- elementwise kernels ----------------
__global__ void norm_kernel(float* __restrict__ buf, const float* __restrict__ a, int total, int K) {
  int i = blockIdx.x * 256 + threadIdx.x;
  if (i >= total) return;
  const int n = i % K;
  float* p = buf + (size_t)i * 8;
  float v[8];
#pragma unroll
  for (int q = 0; q < 8; ++q) v[q] = p[q];
  const float qq[4] = { v[0]*v[0], v[1]*v[1]+v[2]*v[2]+v[3]*v[3],
                        v[4]*v[4]+v[5]*v[5]+v[6]*v[6], v[7]*v[7] };
  float sc[4];
#pragma unroll
  for (int g = 0; g < 4; ++g) {
    const float nr = sqrtf(qq[g]);
    const float sa = sigmoidf(a[n * 4 + g]);
    sc[g] = 1.f / (sa * (nr - 1.f) + 1.f + EPSN);
  }
#pragma unroll
  for (int q = 0; q < 8; ++q) p[q] = v[q] * sc[GRADE8[q]];
}

// MVSiLU; emits f32 (interleaved) + bf16 (blade-major) activations
__global__ void silu_kernel(const float* __restrict__ src, const float* __restrict__ a,
                            const float* __restrict__ b, float* __restrict__ dst,
                            unsigned short* __restrict__ dstbf, int total, int K) {
  int i = blockIdx.x * 256 + threadIdx.x;
  if (i >= total) return;
  const int n = i % K;
  const float* p = src + (size_t)i * 8;
  float v[8];
#pragma unroll
  for (int q = 0; q < 8; ++q) v[q] = p[q];
  const float inv[4] = { v[0], v[1]*v[1]+v[2]*v[2]+v[3]*v[3],
                         v[4]*v[4]+v[5]*v[5]+v[6]*v[6], v[7]*v[7] };
  float gt[4];
#pragma unroll
  for (int g = 0; g < 4; ++g) gt[g] = sigmoidf(a[n*4+g] * inv[g] + b[n*4+g]);
#pragma unroll
  for (int q = 0; q < 8; ++q) {
    const float o = gt[GRADE8[q]] * v[q];
    dst  [(size_t)i*8 + q]     = o;
    dstbf[(size_t)q*total + i] = f2bf(o);
  }
}

// fused: channel-wise GP + bias + /sqrt2 + MVSiLU; bf16 out is blade-major
__global__ void cw_silu_kernel(const float* __restrict__ lin, const float* __restrict__ h,
                               const float* __restrict__ hr, const float* __restrict__ wg,
                               const float* __restrict__ bias, const float* __restrict__ aa,
                               const float* __restrict__ ab, float* __restrict__ dst,
                               unsigned short* __restrict__ dstbf, int total, int K) {
  int i = blockIdx.x * 256 + threadIdx.x;
  if (i >= total) return;
  const int n = i % K;
  const float* pl = lin + (size_t)i * 8;
  const float* ph = h   + (size_t)i * 8;
  const float* pr = hr  + (size_t)i * 8;
  float h8[8], r8[8], o[8];
#pragma unroll
  for (int q = 0; q < 8; ++q) { h8[q] = ph[q]; r8[q] = pr[q]; }
  for (int j = 0; j < 8; ++j) {
    float gp = 0.f;
    const int ns = GP_NSLOT[j];
    for (int s = 0; s < ns; ++s)
      gp += wg[n * NPATH + GP_T[j][s]] * gp_slot_val(j, s, h8, r8);
    o[j] = (pl[j] + (j == 0 ? bias[n] : 0.f) + gp) * INV_SQRT2;
  }
  const float inv[4] = { o[0], o[1]*o[1]+o[2]*o[2]+o[3]*o[3],
                         o[4]*o[4]+o[5]*o[5]+o[6]*o[6], o[7]*o[7] };
  float gt[4];
#pragma unroll
  for (int g = 0; g < 4; ++g) gt[g] = sigmoidf(aa[n*4+g] * inv[g] + ab[n*4+g]);
#pragma unroll
  for (int q = 0; q < 8; ++q) {
    const float v = gt[GRADE8[q]] * o[q];
    dst  [(size_t)i*8 + q]     = v;
    dstbf[(size_t)q*total + i] = f2bf(v);
  }
}

// ---------------- WMMA GEMM: per-blade mv_linear (async double-buffered) ----
// act blade-major [8][B][K] bf16; WT tiled [8][K/32][M][32] bf16; out[b,m,j] f32
__global__ __launch_bounds__(256) void mvlin_gemm_kernel(
    const unsigned short* __restrict__ act, const unsigned short* __restrict__ WT,
    float* __restrict__ out, int K, int M)
{
  __shared__ __align__(16) unsigned short As[2][64 * LDSA];
  __shared__ __align__(16) unsigned short Bs[2][128 * 32];
  const int j = blockIdx.z;
  const int bTile = blockIdx.y * 64;
  const int mTile = blockIdx.x * 128;
  const unsigned short* actJ = act + (size_t)j * BB * K;
  const unsigned short* WTj  = WT  + (size_t)j * K * M;
  const int tid = threadIdx.x;
  const int lane = tid & 31, wv = tid >> 5;
  const int wr = wv >> 2, wc = wv & 3;           // 2x4 wave grid, 32x32 per wave
  const int l2 = lane & 15, hi = lane >> 4;
  // copy coordinates: A = 64x32 (one 16B chunk/thread); B = 128x32 (two 16B chunks)
  const int arow = tid >> 2, acol = (tid & 3) << 3;
  const int bg = tid & 3;
  const int bm0 = tid >> 2, bm1 = 64 + (tid >> 2);
  const unsigned ldsA0 = lds_addr_of(&As[0][arow * LDSA + acol]);
  const unsigned ldsA1 = lds_addr_of(&As[1][arow * LDSA + acol]);
  const unsigned db0 = (unsigned)((bm0 * 32 + ((bg ^ ((bm0 >> 2) & 3)) << 3)) * 2);
  const unsigned db1 = (unsigned)((bm1 * 32 + ((bg ^ ((bm1 >> 2) & 3)) << 3)) * 2);
  const unsigned ldsB_base0 = lds_addr_of(&Bs[0][0]);
  const unsigned ldsB_base1 = lds_addr_of(&Bs[1][0]);

  auto issue_tile = [&](int kk, int buf) {
    const unsigned short* gA = actJ + (size_t)(bTile + arow) * K + (kk << 5) + acol;
    const unsigned short* gB = WTj + ((size_t)kk * M + mTile) * 32;
    const unsigned la = buf ? ldsA1 : ldsA0;
    const unsigned lb = buf ? ldsB_base1 : ldsB_base0;
    async_copy16(la, gA);
    async_copy16(lb + db0, gB + (size_t)tid * 8);
    async_copy16(lb + db1, gB + (size_t)(tid + 256) * 8);
  };

  const v8f vz = {0.f,0.f,0.f,0.f,0.f,0.f,0.f,0.f};
  v8f acc[2][2] = {{vz, vz}, {vz, vz}};
  const int nK = K >> 5;
  issue_tile(0, 0);
  for (int kk = 0; kk < nK; ++kk) {
    const int cur = kk & 1;
    wait_async0();       // my copies into buf[cur] complete
    __syncthreads();     // everyone's copies done; prior reads of buf[cur^1] done
    if (kk + 1 < nK) issue_tile(kk + 1, cur ^ 1);   // overlaps with WMMAs below
    const v16bf af0 = load_frag_a(As[cur], wr * 32 + 0,  l2, hi);
    const v16bf af1 = load_frag_a(As[cur], wr * 32 + 16, l2, hi);
    const v16bf bf0 = load_frag_b(Bs[cur], wc * 32 + 0,  l2, hi);
    const v16bf bf1 = load_frag_b(Bs[cur], wc * 32 + 16, l2, hi);
    acc[0][0] = __builtin_amdgcn_wmma_f32_16x16x32_bf16(false, af0, false, bf0, (short)0, acc[0][0], false, false);
    acc[0][1] = __builtin_amdgcn_wmma_f32_16x16x32_bf16(false, af0, false, bf1, (short)0, acc[0][1], false, false);
    acc[1][0] = __builtin_amdgcn_wmma_f32_16x16x32_bf16(false, af1, false, bf0, (short)0, acc[1][0], false, false);
    acc[1][1] = __builtin_amdgcn_wmma_f32_16x16x32_bf16(false, af1, false, bf1, (short)0, acc[1][1], false, false);
  }
#pragma unroll
  for (int si = 0; si < 2; ++si)
#pragma unroll
    for (int sj = 0; sj < 2; ++sj)
#pragma unroll
      for (int v = 0; v < 8; ++v) {
        const int bI = bTile + wr * 32 + si * 16 + v + 8 * hi;
        const int mI = mTile + wc * 32 + sj * 16 + l2;
        out[((size_t)bI * M + mI) * 8 + j] = acc[si][sj][v];
      }
}

// ---------------- WMMA GEMM: fused (mv_linear + fc_geometric_product) ----
// K' = 8*K (8 slots per channel). A built on the fly from (xL, xR); W prepacked
// tiled [8][K'/32][M][32]. Epilogue: (+bias on blade0) * 1/sqrt2.
__global__ __launch_bounds__(256) void fcgp_gemm_kernel(
    const float* __restrict__ xL, const float* __restrict__ xR,
    const unsigned short* __restrict__ WT, const float* __restrict__ bias,
    float* __restrict__ out, int K, int M)
{
  __shared__ __align__(16) unsigned short As[2][64 * LDSA];
  __shared__ __align__(16) unsigned short Bs[2][128 * 32];
  const int j = blockIdx.z;
  const int bTile = blockIdx.y * 64;
  const int mTile = blockIdx.x * 128;
  const int Kp = K << 3;
  const unsigned short* WTj = WT + (size_t)j * Kp * M;
  const int tid = threadIdx.x;
  const int lane = tid & 31, wv = tid >> 5;
  const int wr = wv >> 2, wc = wv & 3;
  const int l2 = lane & 15, hi = lane >> 4;
  const int rA = tid & 63, ncA = tid >> 6;       // A builder: 64 rows x 4 channels
  const int bg = tid & 3;
  const int bm0 = tid >> 2, bm1 = 64 + (tid >> 2);
  const unsigned db0 = (unsigned)((bm0 * 32 + ((bg ^ ((bm0 >> 2) & 3)) << 3)) * 2);
  const unsigned db1 = (unsigned)((bm1 * 32 + ((bg ^ ((bm1 >> 2) & 3)) << 3)) * 2);
  const unsigned ldsB_base0 = lds_addr_of(&Bs[0][0]);
  const unsigned ldsB_base1 = lds_addr_of(&Bs[1][0]);
  const int nsl = GP_NSLOT[j];

  auto issue_B = [&](int kk, int buf) {
    const unsigned short* gB = WTj + ((size_t)kk * M + mTile) * 32;
    const unsigned lb = buf ? ldsB_base1 : ldsB_base0;
    async_copy16(lb + db0, gB + (size_t)tid * 8);
    async_copy16(lb + db1, gB + (size_t)(tid + 256) * 8);
  };
  auto build_store_A = [&](const float* a8, const float* r8, int buf) {
    v8us sl;
    for (int s = 0; s < 8; ++s) {
      float v = 0.f;
      if (s < nsl)       v = gp_slot_val(j, s, a8, r8);
      else if (s == nsl) v = a8[j];
      sl[s] = f2bf(v);
    }
    *(v8us*)(&As[buf][rA * LDSA + (ncA << 3)]) = sl;
  };

  const v8f vz = {0.f,0.f,0.f,0.f,0.f,0.f,0.f,0.f};
  v8f acc[2][2] = {{vz, vz}, {vz, vz}};
  const int nK = Kp >> 5;
  // prologue: tile 0
  issue_B(0, 0);
  {
    const int b = bTile + rA, n = ncA;
    const float* pa = xL + ((size_t)b * K + n) * 8;
    const float* pr = xR + ((size_t)b * K + n) * 8;
    const v4f a0 = *(const v4f*)pa, a1 = *(const v4f*)(pa + 4);
    const v4f r0 = *(const v4f*)pr, r1 = *(const v4f*)(pr + 4);
    const float a8[8] = {a0[0],a0[1],a0[2],a0[3],a1[0],a1[1],a1[2],a1[3]};
    const float r8[8] = {r0[0],r0[1],r0[2],r0[3],r1[0],r1[1],r1[2],r1[3]};
    build_store_A(a8, r8, 0);
  }
  for (int kk = 0; kk < nK; ++kk) {
    const int cur = kk & 1;
    const bool more = (kk + 1 < nK);
    v4f a0, a1, r0, r1;
    if (more) {  // preload next tile's x/xr (latency hidden behind WMMAs)
      const int b = bTile + rA, n = ((kk + 1) << 2) + ncA;
      const float* pa = xL + ((size_t)b * K + n) * 8;
      const float* pr = xR + ((size_t)b * K + n) * 8;
      a0 = *(const v4f*)pa; a1 = *(const v4f*)(pa + 4);
      r0 = *(const v4f*)pr; r1 = *(const v4f*)(pr + 4);
    }
    wait_async0();
    __syncthreads();
    if (more) issue_B(kk + 1, cur ^ 1);
    const v16bf af0 = load_frag_a(As[cur], wr * 32 + 0,  l2, hi);
    const v16bf af1 = load_frag_a(As[cur], wr * 32 + 16, l2, hi);
    const v16bf bf0 = load_frag_b(Bs[cur], wc * 32 + 0,  l2, hi);
    const v16bf bf1 = load_frag_b(Bs[cur], wc * 32 + 16, l2, hi);
    acc[0][0] = __builtin_amdgcn_wmma_f32_16x16x32_bf16(false, af0, false, bf0, (short)0, acc[0][0], false, false);
    acc[0][1] = __builtin_amdgcn_wmma_f32_16x16x32_bf16(false, af0, false, bf1, (short)0, acc[0][1], false, false);
    acc[1][0] = __builtin_amdgcn_wmma_f32_16x16x32_bf16(false, af1, false, bf0, (short)0, acc[1][0], false, false);
    acc[1][1] = __builtin_amdgcn_wmma_f32_16x16x32_bf16(false, af1, false, bf1, (short)0, acc[1][1], false, false);
    if (more) {
      const float a8[8] = {a0[0],a0[1],a0[2],a0[3],a1[0],a1[1],a1[2],a1[3]};
      const float r8[8] = {r0[0],r0[1],r0[2],r0[3],r1[0],r1[1],r1[2],r1[3]};
      build_store_A(a8, r8, cur ^ 1);
    }
  }
#pragma unroll
  for (int si = 0; si < 2; ++si)
#pragma unroll
    for (int sj = 0; sj < 2; ++sj)
#pragma unroll
      for (int v = 0; v < 8; ++v) {
        const int bI = bTile + wr * 32 + si * 16 + v + 8 * hi;
        const int mI = mTile + wc * 32 + sj * 16 + l2;
        float r = acc[si][sj][v];
        if (j == 0) r += bias[mI];
        out[((size_t)bI * M + mI) * 8 + j] = r * INV_SQRT2;
      }
}

// ---------------- host orchestration ----------------
extern "C" void kernel_launch(void* const* d_in, const int* in_sizes, int n_in,
                              void* d_out, int out_size, void* d_ws, size_t ws_size,
                              hipStream_t stream) {
  (void)in_sizes; (void)n_in; (void)out_size; (void)ws_size;
  const float* x     = (const float*)d_in[0];
  const float* lr1_w = (const float*)d_in[1];
  const float* n1_a  = (const float*)d_in[2];
  const float* ll1_w = (const float*)d_in[3];
  const float* ll1_b = (const float*)d_in[4];
  const float* w1    = (const float*)d_in[5];
  const float* act_a = (const float*)d_in[6];
  const float* act_b = (const float*)d_in[7];
  const float* lrg_w = (const float*)d_in[8];
  const float* ng_a  = (const float*)d_in[9];
  const float* llg_w = (const float*)d_in[10];
  const float* llg_b = (const float*)d_in[11];
  const float* wg    = (const float*)d_in[12];
  const float* lr2_w = (const float*)d_in[13];
  const float* n2_a  = (const float*)d_in[14];
  const float* ll2_w = (const float*)d_in[15];
  const float* ll2_b = (const float*)d_in[16];
  const float* w2    = (const float*)d_in[17];
  float* out = (float*)d_out;

  char* base = (char*)d_ws;
  size_t off = 0;
  auto take = [&](size_t bytes) { size_t o = off; off += (bytes + 255) & ~(size_t)255; return o; };
  unsigned short* xbf  = (unsigned short*)(base + take((size_t)8 * BB * CIN  * 2));
  unsigned short* hbf  = (unsigned short*)(base + take((size_t)8 * BB * CHID * 2));
  unsigned short* h2bf = (unsigned short*)(base + take((size_t)8 * BB * CHID * 2));
  float* t0 = (float*)(base + take((size_t)BB * CHID * 8 * 4));   // xr / hr
  float* t1 = (float*)(base + take((size_t)BB * CHID * 8 * 4));   // pre-activation
  float* hA = (float*)(base + take((size_t)BB * CHID * 8 * 4));   // layer1 activation
  float* h2 = (float*)(base + take((size_t)BB * CHID * 8 * 4));   // cw-layer activation
  unsigned short* wl1  = (unsigned short*)(base + take((size_t)8 * CIN  * CIN  * 2));
  unsigned short* wlg  = (unsigned short*)(base + take((size_t)8 * CHID * CHID * 2));
  unsigned short* wllg = (unsigned short*)(base + take((size_t)8 * CHID * CHID * 2));
  unsigned short* wl2  = (unsigned short*)(base + take((size_t)8 * CHID * CHID * 2));
  unsigned short* wp1  = (unsigned short*)(base + take((size_t)8 * (CIN  * 8) * CHID * 2));
  unsigned short* wp2  = (unsigned short*)(base + take((size_t)8 * (CHID * 8) * COUT * 2));

  const dim3 blk(256);
  auto g1 = [](int n) { return dim3((unsigned)((n + 255) / 256)); };

  // ---- prepack (deterministic, every call) ----
  { int n = 8 * BB * CIN;        pack_bf16_bm_kernel<<<g1(n), blk, 0, stream>>>(x, xbf, BB * CIN); }
  { int n = 8 * CIN  * CIN;      pack_wlin_kernel<<<g1(n), blk, 0, stream>>>(lr1_w, wl1,  CIN,  CIN); }
  { int n = 8 * CHID * CHID;     pack_wlin_kernel<<<g1(n), blk, 0, stream>>>(lrg_w, wlg,  CHID, CHID); }
  { int n = 8 * CHID * CHID;     pack_wlin_kernel<<<g1(n), blk, 0, stream>>>(llg_w, wllg, CHID, CHID); }
  { int n = 8 * CHID * CHID;     pack_wlin_kernel<<<g1(n), blk, 0, stream>>>(lr2_w, wl2,  CHID, CHID); }
  { int n = 8 * CIN  * 8 * CHID; pack_wp_kernel  <<<g1(n), blk, 0, stream>>>(ll1_w, w1, wp1, CIN,  CHID); }
  { int n = 8 * CHID * 8 * COUT; pack_wp_kernel  <<<g1(n), blk, 0, stream>>>(ll2_w, w2, wp2, CHID, COUT); }

  // ---- layer 1: fcgp(IN -> HID) ----
  mvlin_gemm_kernel<<<dim3(CIN / 128, BB / 64, 8), blk, 0, stream>>>(xbf, wl1, t0, CIN, CIN);
  { int n = BB * CIN;  norm_kernel<<<g1(n), blk, 0, stream>>>(t0, n1_a, n, CIN); }
  fcgp_gemm_kernel<<<dim3(CHID / 128, BB / 64, 8), blk, 0, stream>>>(x, t0, wp1, ll1_b, t1, CIN, CHID);
  { int n = BB * CHID; silu_kernel<<<g1(n), blk, 0, stream>>>(t1, act_a, act_b, hA, hbf, n, CHID); }

  // ---- channel-wise steerable GP layer (HID) ----
  mvlin_gemm_kernel<<<dim3(CHID / 128, BB / 64, 8), blk, 0, stream>>>(hbf, wlg, t0, CHID, CHID);
  { int n = BB * CHID; norm_kernel<<<g1(n), blk, 0, stream>>>(t0, ng_a, n, CHID); }
  mvlin_gemm_kernel<<<dim3(CHID / 128, BB / 64, 8), blk, 0, stream>>>(hbf, wllg, t1, CHID, CHID);
  { int n = BB * CHID; cw_silu_kernel<<<g1(n), blk, 0, stream>>>(t1, hA, t0, wg, llg_b, act_a, act_b, h2, h2bf, n, CHID); }

  // ---- layer 2: fcgp(HID -> OUT), writes d_out ----
  mvlin_gemm_kernel<<<dim3(CHID / 128, BB / 64, 8), blk, 0, stream>>>(h2bf, wl2, t0, CHID, CHID);
  { int n = BB * CHID; norm_kernel<<<g1(n), blk, 0, stream>>>(t0, n2_a, n, CHID); }
  fcgp_gemm_kernel<<<dim3(COUT / 128, BB / 64, 8), blk, 0, stream>>>(h2, t0, wp2, ll2_b, out, CHID, COUT);
}
